// SharpeOptimizedLoss_15702400434455
// MI455X (gfx1250) — compile-verified, matrix operand order checked by reading, not run on hardware
//
#include <hip/hip_runtime.h>
#include <math.h>

// ---------------------------------------------------------------------------
// SharpeOptimizedLoss for MI455X (gfx1250, wave32).
// N = 2^21 floats (16 MB inputs) -> entirely L2-resident; pass-count bound.
// Pipeline:
//   k_init    : zero radix histograms + select scratch
//   k_rs      : s = tanh(pred), r = ret*s          (materialize once)
//   k_main    : fused per-block reduction: sums, |r|, |dr|, |ds|, rolling vol,
//               drawdown monoid, and WMMA-iu8 indicator counts (pos / sign pairs)
//   k_hist/k_select x3 : exact 32-bit radix select of the k-th smallest r
//   k_cvsum   : sum of r strictly below threshold (per-block partials)
//   k_final   : deterministic tree combine + scalar loss
// ---------------------------------------------------------------------------

#define NTHREADS 256
#define CHUNK    4096
#define EPT      (CHUNK / NTHREADS)   // 16 elements per thread (contiguous)
#define WIN      20
#define NHISTGRID 512
#define FLT_BIG  3.402823466e+38f

typedef int v8i __attribute__((ext_vector_type(8)));

struct Acc {
  float sum_r, sum_r2, sum_abs, sum_absdr, sum_absds, sum_rv, sum_rv2;
  float t, maxP, minP, maxDD;          // drawdown monoid
  int   pos16, same16, opp16;          // 16x indicator counts (WMMA output)
};

__device__ inline Acc acc_identity() {
  Acc a;
  a.sum_r = a.sum_r2 = a.sum_abs = a.sum_absdr = a.sum_absds = 0.f;
  a.sum_rv = a.sum_rv2 = 0.f;
  a.t = 0.f; a.maxP = -FLT_BIG; a.minP = FLT_BIG; a.maxDD = 0.f;
  a.pos16 = a.same16 = a.opp16 = 0;
  return a;
}

// Order-sensitive associative combine (a precedes b in sequence order).
__device__ inline Acc acc_combine(const Acc& a, const Acc& b) {
  Acc c;
  c.sum_r     = a.sum_r     + b.sum_r;
  c.sum_r2    = a.sum_r2    + b.sum_r2;
  c.sum_abs   = a.sum_abs   + b.sum_abs;
  c.sum_absdr = a.sum_absdr + b.sum_absdr;
  c.sum_absds = a.sum_absds + b.sum_absds;
  c.sum_rv    = a.sum_rv    + b.sum_rv;
  c.sum_rv2   = a.sum_rv2   + b.sum_rv2;
  c.maxP  = fmaxf(a.maxP, a.t + b.maxP);
  c.minP  = fminf(a.minP, a.t + b.minP);
  c.maxDD = fmaxf(fmaxf(a.maxDD, b.maxDD), a.maxP - (a.t + b.minP));
  c.t     = a.t + b.t;
  c.pos16  = a.pos16  + b.pos16;
  c.same16 = a.same16 + b.same16;
  c.opp16  = a.opp16  + b.opp16;
  return c;
}

// Monotonic float -> uint key (ascending key order == ascending float order)
__device__ inline unsigned keyOf(float x) {
  unsigned u = __float_as_uint(x);
  return (u & 0x80000000u) ? ~u : (u | 0x80000000u);
}
__device__ inline float invKey(unsigned k) {
  return (k & 0x80000000u) ? __uint_as_float(k & 0x7FFFFFFFu)
                           : __uint_as_float(~k);
}

// ---------------------------------------------------------------------------
__global__ __launch_bounds__(NTHREADS)
void k_init(unsigned* __restrict__ hist3x, unsigned* __restrict__ sel) {
  int i = blockIdx.x * NTHREADS + threadIdx.x;
  if (i < 3 * 2048) hist3x[i] = 0u;
  if (i < 8) sel[i] = 0u;
}

__global__ __launch_bounds__(NTHREADS)
void k_rs(const float* __restrict__ pred, const float* __restrict__ rets,
          float* __restrict__ rArr, float* __restrict__ sArr, int N) {
  int i = blockIdx.x * NTHREADS + threadIdx.x;
  if (i < N) {
    float s = tanhf(pred[i]);
    sArr[i] = s;
    rArr[i] = rets[i] * s;
  }
}

// ---------------------------------------------------------------------------
__global__ __launch_bounds__(NTHREADS)
void k_main(const float* __restrict__ rArr, const float* __restrict__ sArr,
            Acc* __restrict__ rec, int N) {
  __shared__ float sr[CHUNK + 32];
  __shared__ float ss[CHUNK + 32];
  __shared__ Acc   red[NTHREADS];

  const int  tid  = threadIdx.x;
  const long base = (long)blockIdx.x * CHUNK;

  for (int idx = tid; idx < CHUNK + 32; idx += NTHREADS) {
    long g = base + idx;
    bool ok = g < (long)N;
    sr[idx] = ok ? rArr[g] : 0.0f;
    ss[idx] = ok ? sArr[g] : 0.0f;
  }
  __syncthreads();

  Acc a = acc_identity();
  unsigned posW[8]  = {0,0,0,0,0,0,0,0};
  unsigned sameW[8] = {0,0,0,0,0,0,0,0};
  unsigned oppW[8]  = {0,0,0,0,0,0,0,0};

  const int l0 = tid * EPT;
  for (int j = 0; j < EPT; ++j) {
    int  li = l0 + j;
    long gi = base + li;
    float r = sr[li];

    a.sum_r  += r;
    a.sum_r2 += r * r;
    a.sum_abs += fabsf(r);
    if (r > 0.0f) posW[j >> 2] |= 1u << ((j & 3) * 8);

    if (gi < (long)N - 1) {
      float rn = sr[li + 1];
      a.sum_absdr += fabsf(rn - r);
      a.sum_absds += fabsf(ss[li + 1] - ss[li]);
      int sa = (r  > 0.0f) - (r  < 0.0f);
      int sb = (rn > 0.0f) - (rn < 0.0f);
      int pr = sa * sb;
      if (pr > 0)      sameW[j >> 2] |= 1u << ((j & 3) * 8);
      else if (pr < 0) oppW[j >> 2]  |= 1u << ((j & 3) * 8);
    }

    if (gi < (long)N - WIN) {   // rolling window k = gi, k in [0, N-WIN)
      float wsum = 0.f, wsq = 0.f;
      #pragma unroll
      for (int w = 0; w < WIN; ++w) {
        float x = sr[li + w];
        wsum += x; wsq += x * x;
      }
      float m   = wsum * (1.0f / (float)WIN);
      float var = (wsq - wsum * m) * (1.0f / (float)(WIN - 1));
      float rv  = sqrtf(fmaxf(var, 0.0f));
      a.sum_rv  += rv;
      a.sum_rv2 += rv * rv;
    }

    // sequential drawdown monoid over this thread's contiguous segment
    a.t    += r;
    a.maxP  = fmaxf(a.maxP, a.t);
    a.minP  = fminf(a.minP, a.t);
    a.maxDD = fmaxf(a.maxDD, a.maxP - a.t);
  }

  // --- Exact integer indicator reduction on the matrix unit ---------------
  // Each wave's 32 lanes supply the 16x64 u8 A matrix (1024 indicator bytes,
  // any layout). B = all-ones -> sum over all D entries == 16 * total count.
  v8i A, S, O, Bones, Cz, D;
  #pragma unroll
  for (int c = 0; c < 8; ++c) {
    A[c] = (int)posW[c];
    S[c] = (int)sameW[c];
    O[c] = (int)oppW[c];
    Bones[c] = 0x01010101;
    Cz[c] = 0;
  }
  D = __builtin_amdgcn_wmma_i32_16x16x64_iu8(false, A, false, Bones, Cz, false, false);
  a.pos16  = D[0]+D[1]+D[2]+D[3]+D[4]+D[5]+D[6]+D[7];
  D = __builtin_amdgcn_wmma_i32_16x16x64_iu8(false, S, false, Bones, Cz, false, false);
  a.same16 = D[0]+D[1]+D[2]+D[3]+D[4]+D[5]+D[6]+D[7];
  D = __builtin_amdgcn_wmma_i32_16x16x64_iu8(false, O, false, Bones, Cz, false, false);
  a.opp16  = D[0]+D[1]+D[2]+D[3]+D[4]+D[5]+D[6]+D[7];

  // --- Order-preserving LDS tree combine ----------------------------------
  red[tid] = a;
  __syncthreads();
  for (int s = 1; s < NTHREADS; s <<= 1) {
    if ((tid & (2 * s - 1)) == 0)
      red[tid] = acc_combine(red[tid], red[tid + s]);
    __syncthreads();
  }
  if (tid == 0) rec[blockIdx.x] = red[0];
}

// ---------------------------------------------------------------------------
// Radix-select histogram pass. mode 0: bits[31:21]; mode 1: bits[20:10]
// filtered by prefix; mode 2: bits[9:0] filtered by 22-bit prefix.
__global__ __launch_bounds__(NTHREADS)
void k_hist(const float* __restrict__ rArr, int N, unsigned* __restrict__ hist,
            const unsigned* __restrict__ sel, int mode) {
  __shared__ unsigned h[2048];
  const int tid = threadIdx.x;
  for (int i = tid; i < 2048; i += NTHREADS) h[i] = 0u;
  __syncthreads();
  unsigned p = (mode == 0) ? 0u : sel[0];
  for (long i = (long)blockIdx.x * NTHREADS + tid; i < (long)N;
       i += (long)gridDim.x * NTHREADS) {
    unsigned k = keyOf(rArr[i]);
    if (mode == 0) {
      atomicAdd(&h[k >> 21], 1u);
    } else if (mode == 1) {
      if ((k >> 21) == p) atomicAdd(&h[(k >> 10) & 2047u], 1u);
    } else {
      if ((k >> 10) == p) atomicAdd(&h[k & 1023u], 1u);
    }
  }
  __syncthreads();
  int nb = (mode == 2) ? 1024 : 2048;
  for (int i = tid; i < nb; i += NTHREADS)
    if (h[i]) atomicAdd(&hist[i], h[i]);
}

__global__ void k_select(const unsigned* __restrict__ hist,
                         unsigned* __restrict__ sel, int mode, int kcv) {
  unsigned cb = (mode == 0) ? 0u : sel[1];
  unsigned p  = (mode == 0) ? 0u : sel[0];
  int nb = (mode == 2) ? 1024 : 2048;
  unsigned b = 0;
  for (int i = 0; i < nb; ++i) {
    unsigned c = hist[i];
    if (cb + c >= (unsigned)kcv) { b = (unsigned)i; break; }
    cb += c;
  }
  if (mode == 0)      { sel[0] = b;                sel[1] = cb; }
  else if (mode == 1) { sel[0] = (p << 11) | b;    sel[1] = cb; }
  else                { sel[2] = (p << 10) | b;    sel[3] = cb; }
}

__global__ __launch_bounds__(NTHREADS)
void k_cvsum(const float* __restrict__ rArr, int N,
             const unsigned* __restrict__ sel, float* __restrict__ cvPart) {
  __shared__ float red[NTHREADS];
  const int tid = threadIdx.x;
  unsigned T = sel[2];
  float s = 0.0f;
  for (long i = (long)blockIdx.x * NTHREADS + tid; i < (long)N;
       i += (long)gridDim.x * NTHREADS) {
    float r = rArr[i];
    if (keyOf(r) < T) s += r;
  }
  red[tid] = s;
  __syncthreads();
  for (int st = NTHREADS / 2; st > 0; st >>= 1) {
    if (tid < st) red[tid] += red[tid + st];
    __syncthreads();
  }
  if (tid == 0) cvPart[blockIdx.x] = red[0];
}

// ---------------------------------------------------------------------------
__global__ __launch_bounds__(NTHREADS)
void k_final(const Acc* __restrict__ rec, int nRec,
             const float* __restrict__ cvPart, int nCv,
             const unsigned* __restrict__ sel,
             float* __restrict__ out, int N, int kcv) {
  __shared__ Acc   red[NTHREADS];
  __shared__ float cred[NTHREADS];
  const int t = threadIdx.x;

  Acc a = (2 * t     < nRec) ? rec[2 * t]     : acc_identity();
  Acc b = (2 * t + 1 < nRec) ? rec[2 * t + 1] : acc_identity();
  red[t] = acc_combine(a, b);       // preserves sequence order

  float c = 0.0f;
  for (int i = t; i < nCv; i += NTHREADS) c += cvPart[i];
  cred[t] = c;
  __syncthreads();

  for (int s = 1; s < NTHREADS; s <<= 1) {
    if ((t & (2 * s - 1)) == 0) {
      red[t]  = acc_combine(red[t], red[t + s]);
      cred[t] = cred[t] + cred[t + s];
    }
    __syncthreads();
  }

  if (t == 0) {
    Acc A = red[0];
    double n = (double)N;

    double mean = (double)A.sum_r / n;
    double var  = ((double)A.sum_r2 - (double)A.sum_r * A.sum_r / n) / (n - 1.0);
    double stdv = sqrt(fmax(var, 0.0)) + 1e-8;
    double base_sharpe = mean / stdv;

    double nw = n - (double)WIN;
    double varrv = ((double)A.sum_rv2 - (double)A.sum_rv * A.sum_rv / nw) / (nw - 1.0);
    double vol_st = 1.0 / (sqrt(fmax(varrv, 0.0)) + 1e-6);
    double enh = base_sharpe * (1.0 + 0.1 * vol_st);

    double rmag = (double)A.sum_abs / n;
    double posf = ((double)A.pos16 / 16.0) / n;
    double mom  = ((double)(A.same16 - A.opp16) / 16.0) / (n - 1.0);
    double rsm  = 1.0 / ((double)A.sum_absdr / (n - 1.0) + 1e-6);
    double scg  = (double)A.sum_absds / (n - 1.0);
    double ssm  = 1.0 / (scg + 1e-6);
    double ddpen = fmax((double)A.maxDD - 0.05, 0.0);

    unsigned T  = sel[2];
    unsigned cl = sel[3];
    double tv = (double)invKey(T);
    double cvar = -((double)cred[0] + (double)(kcv - (int)cl) * tv) / (double)kcv;

    double sharpe_c = 0.4 * enh + 0.25 * rmag + 0.15 * posf +
                      0.1 * mom + 0.05 * rsm + 0.05 * ssm;
    double risk = 0.05 * cvar + 0.02 * ddpen + 0.01 * scg;

    out[0] = (float)(-(0.6 * rmag + 0.4 * sharpe_c - risk));
  }
}

// ---------------------------------------------------------------------------
extern "C" void kernel_launch(void* const* d_in, const int* in_sizes, int n_in,
                              void* d_out, int out_size, void* d_ws, size_t ws_size,
                              hipStream_t stream) {
  const float* pred = (const float*)d_in[0];
  const float* rets = (const float*)d_in[1];
  int N = in_sizes[1];                     // returns: [N]
  float* out = (float*)d_out;

  // workspace layout
  char* ws = (char*)d_ws;
  float* rArr = (float*)ws;
  float* sArr = rArr + N;
  Acc*   rec  = (Acc*)(sArr + N);
  unsigned* hist  = (unsigned*)((char*)rec + (size_t)512 * sizeof(Acc));
  unsigned* hist1 = hist;
  unsigned* hist2 = hist + 2048;
  unsigned* hist3 = hist + 4096;
  unsigned* sel   = hist + 6144;           // 8 uints
  float* cvPart   = (float*)(sel + 8);     // NHISTGRID floats

  int kcv = (int)((double)N * 0.03);
  if (kcv < 1) kcv = 1;
  int gMain = (N + CHUNK - 1) / CHUNK;     // 512 for N = 2^21

  hipLaunchKernelGGL(k_init, dim3(25), dim3(NTHREADS), 0, stream, hist, sel);
  hipLaunchKernelGGL(k_rs, dim3((N + NTHREADS - 1) / NTHREADS), dim3(NTHREADS),
                     0, stream, pred, rets, rArr, sArr, N);
  hipLaunchKernelGGL(k_main, dim3(gMain), dim3(NTHREADS), 0, stream,
                     rArr, sArr, rec, N);

  hipLaunchKernelGGL(k_hist, dim3(NHISTGRID), dim3(NTHREADS), 0, stream,
                     rArr, N, hist1, sel, 0);
  hipLaunchKernelGGL(k_select, dim3(1), dim3(1), 0, stream, hist1, sel, 0, kcv);
  hipLaunchKernelGGL(k_hist, dim3(NHISTGRID), dim3(NTHREADS), 0, stream,
                     rArr, N, hist2, sel, 1);
  hipLaunchKernelGGL(k_select, dim3(1), dim3(1), 0, stream, hist2, sel, 1, kcv);
  hipLaunchKernelGGL(k_hist, dim3(NHISTGRID), dim3(NTHREADS), 0, stream,
                     rArr, N, hist3, sel, 2);
  hipLaunchKernelGGL(k_select, dim3(1), dim3(1), 0, stream, hist3, sel, 2, kcv);

  hipLaunchKernelGGL(k_cvsum, dim3(NHISTGRID), dim3(NTHREADS), 0, stream,
                     rArr, N, sel, cvPart);
  hipLaunchKernelGGL(k_final, dim3(1), dim3(NTHREADS), 0, stream,
                     rec, gMain, cvPart, NHISTGRID, sel, out, N, kcv);
}
